// SolventModel_50328426774781
// MI455X (gfx1250) — matrix operand order repaired and checked by reading, not compiled, not run on hardware
//
#include <hip/hip_runtime.h>
#include <stdint.h>

// ---------------- problem constants ----------------
#define GN       256
#define WORDS    (GN * GN * GN / 32)        // 524288 words (2MB per bitmask)
#define SPACING  (100.0f / 256.0f)          // exactly 0.390625
#define SOLV_R   1.1f

typedef __attribute__((ext_vector_type(16))) _Float16 v16h;
typedef __attribute__((ext_vector_type(8)))  float    v8f;

// ---- CDNA5 async global->LDS path (ASYNCcnt), with portable fallback ------
#if defined(__gfx1250__) && __has_builtin(__builtin_amdgcn_global_load_async_to_lds_b32)
#define HAVE_ASYNC_LDS 1
__device__ __forceinline__ void async_copy_b32(const void* g, void* l) {
    __builtin_amdgcn_global_load_async_to_lds_b32(
        (__attribute__((address_space(1))) int*)g,
        (__attribute__((address_space(3))) int*)l,
        /*offset=*/0, /*cpol=*/0);
}
__device__ __forceinline__ void async_wait0() {
#if __has_builtin(__builtin_amdgcn_s_wait_asynccnt)
    __builtin_amdgcn_s_wait_asynccnt(0);
#else
    asm volatile("s_wait_asynccnt 0x0" ::: "memory");
#endif
}
#else
#define HAVE_ASYNC_LDS 0
#endif

// Gaussian 3-tap weights: sigma = SOLV_R/SPACING/4, taps [g,1,g]/(1+2g)
__device__ inline void gauss_w(float& w0, float& w1) {
    const float sigma = SOLV_R / SPACING / 4.0f;
    const float g = __expf(-1.0f / (2.0f * sigma * sigma));
    w1 = 1.0f / (1.0f + 2.0f * g);
    w0 = g * w1;
}

// ---------------------------------------------------------------------------
// Kernel 0: zero the P and B bitmasks (4MB total) with 16B stores
// ---------------------------------------------------------------------------
__global__ void __launch_bounds__(256) k_zero(uint4* p, int n) {
    const int i = blockIdx.x * 256 + threadIdx.x;
    if (i < n) p[i] = make_uint4(0u, 0u, 0u, 0u);
}

// ---------------------------------------------------------------------------
// Kernel 1: atom scatter -> protein / boundary bitmasks (atomicOr)
// One block per atom; 4913 candidate cells in the 17^3 neighborhood.
// ---------------------------------------------------------------------------
__global__ void __launch_bounds__(256)
k_scatter(const float* __restrict__ xyz, const float* __restrict__ vdw,
          uint32_t* __restrict__ P, uint32_t* __restrict__ B, int n_atoms) {
    const int atom = blockIdx.x;
    if (atom >= n_atoms) return;
    if (atom + 1 < n_atoms) __builtin_prefetch(xyz + 3 * (atom + 1), 0, 1);

    const float ax = xyz[atom * 3 + 0];
    const float ay = xyz[atom * 3 + 1];
    const float az = xyz[atom * 3 + 2];
    const float vr   = vdw[atom];
    const float vr2  = vr * vr;
    const float vro  = vr + SOLV_R;
    const float vro2 = vro * vro;

    const int bx = (int)floorf(ax * (1.0f / SPACING));
    const int by = (int)floorf(ay * (1.0f / SPACING));
    const int bz = (int)floorf(az * (1.0f / SPACING));

    for (int t = threadIdx.x; t < 17 * 17 * 17; t += 256) {
        const int oz = t % 17 - 8;
        const int r1 = t / 17;
        const int oy = r1 % 17 - 8;
        const int ox = r1 / 17 - 8;
        const int cx = bx + ox, cy = by + oy, cz = bz + oz;
        const float dx = (float)cx * SPACING - ax;
        const float dy = (float)cy * SPACING - ay;
        const float dz = (float)cz * SPACING - az;
        const float dd = dx * dx + dy * dy + dz * dz;
        if (dd < vro2) {
            const int flat = ((((cx & 255) << 8) | (cy & 255)) << 8) | (cz & 255);
            const uint32_t bit = 1u << (flat & 31);
            if (dd < vr2) atomicOr(&P[flat >> 5], bit);
            else          atomicOr(&B[flat >> 5], bit);
        }
    }
}

// ---------------------------------------------------------------------------
// Kernel 2: B' = B & ~P (in place; P|B is invariant under this, so the
// neighbor recompute below is race-free), def_solv = ~(P|B), plus z-axis
// bit-dilations of def_solv by radius 1 and 2 (z = bit axis, 8 words/row,
// wraparound within the row).
// ---------------------------------------------------------------------------
__global__ void __launch_bounds__(256)
k_finalize(uint32_t* __restrict__ P, uint32_t* __restrict__ B,
           uint32_t* __restrict__ Z0, uint32_t* __restrict__ Z1,
           uint32_t* __restrict__ Z2) {
    const int w  = blockIdx.x * 256 + threadIdx.x;  // 0..WORDS-1
    const int wz = w & 7;
    const int row = w >> 3;                          // (x,y) row of 8 words
    const int wm = (row << 3) | ((wz - 1) & 7);
    const int wp = (row << 3) | ((wz + 1) & 7);

    const uint32_t p = P[w], braw = B[w];
    B[w] = braw & ~p;                                // boundary final

    const uint32_t s0 = ~(p | braw);                 // def_solv (P|B == P|B')
    const uint32_t sm = ~(P[wm] | B[wm]);
    const uint32_t sp = ~(P[wp] | B[wp]);

    const uint32_t d1 = s0
        | (s0 << 1) | (sm >> 31)
        | (s0 >> 1) | (sp << 31);
    const uint32_t d2 = d1
        | (s0 << 2) | (sm >> 30)
        | (s0 >> 2) | (sp << 30);
    Z0[w] = s0; Z1[w] = d1; Z2[w] = d2;
}

// ---------------------------------------------------------------------------
// Kernel 3: has_nb = dilation of def_solv by ball ||o||^2 <= 5, decomposed as
// 21 (dx,dy) offsets, each ORing the z-pre-dilated row of radius
// a = floor(sqrt(5 - dx^2 - dy^2)).
// ---------------------------------------------------------------------------
__global__ void __launch_bounds__(256)
k_dilate(const uint32_t* __restrict__ Z0, const uint32_t* __restrict__ Z1,
         const uint32_t* __restrict__ Z2, uint32_t* __restrict__ H) {
    const int w  = blockIdx.x * 256 + threadIdx.x;
    const int wz = w & 7;
    const int y  = (w >> 3) & 255;
    const int x  = w >> 11;
    const uint32_t* __restrict__ Zs[3] = { Z0, Z1, Z2 };
    uint32_t h = 0;
    #pragma unroll
    for (int dy = -2; dy <= 2; ++dy) {
        #pragma unroll
        for (int dx = -2; dx <= 2; ++dx) {
            const int r2 = dx * dx + dy * dy;
            if (r2 > 5) continue;                     // drops the 4 corners
            const int a = (r2 <= 1) ? 2 : ((r2 <= 4) ? 1 : 0);
            const int idx = (((((x + dx) & 255) << 8) | ((y + dy) & 255)) << 3) | wz;
            h |= Zs[a][idx];
        }
    }
    H[w] = h;
}

// ---------------------------------------------------------------------------
// Kernel 4: blur pass 1 (x axis). solvent = ~(P | (B' & ~H)) computed from
// bits on the fly at x-1,x,x+1; 3-tap conv -> f32 volume in d_out.
// ---------------------------------------------------------------------------
__global__ void __launch_bounds__(256)
k_blur_x(const uint32_t* __restrict__ P, const uint32_t* __restrict__ B,
         const uint32_t* __restrict__ H, float* __restrict__ out) {
    const int w  = blockIdx.x * 256 + threadIdx.x;
    const int wz = w & 7;
    const int y  = (w >> 3) & 255;
    const int x  = w >> 11;

    float w0, w1; gauss_w(w0, w1);

    auto sword = [&](int xx) -> uint32_t {
        const int i = (((xx << 8) | y) << 3) | wz;
        return ~(P[i] | (B[i] & ~H[i]));
    };
    const uint32_t sm = sword((x - 1) & 255);
    const uint32_t s0 = sword(x);
    const uint32_t sp = sword((x + 1) & 255);

    const int obase = w << 5;
    #pragma unroll
    for (int z = 0; z < 32; ++z) {
        const float v = w0 * (float)((sm >> z) & 1u)
                      + w1 * (float)((s0 >> z) & 1u)
                      + w0 * (float)((sp >> z) & 1u);
        out[obase + z] = v;
    }
}

// ---------------------------------------------------------------------------
// WMMA helper: 3-tap wrap conv over a 16-wide tile as D = A(16x32) x B(32x16)
// A: banded Toeplitz [w0,w1,w0]; col 16 = left-halo weight for m=0,
//    col 17 = right-halo weight for m=15; cols 18..31 zero.
// Contiguous k order: element s of lane half h holds k = 16h + s (B matches).
// ---------------------------------------------------------------------------
__device__ inline v16h build_A(float w0, float w1) {
    const int lane = threadIdx.x & 31;
    const int m    = lane & 15;
    const int kadd = (lane >> 4) << 4;               // lanes 16-31: K += 16
    v16h a;
    #pragma unroll
    for (int j = 0; j < 8; ++j) {
        const int kb = (2 * j) + kadd;
        #pragma unroll
        for (int h = 0; h < 2; ++h) {
            const int k = kb + h;
            float v = 0.0f;
            if (k < 16) {
                const int d = k - m;
                v = (d == 0) ? w1 : ((d == -1 || d == 1) ? w0 : 0.0f);
            } else if (k == 16) { v = (m == 0)  ? w0 : 0.0f; }
            else if   (k == 17) { v = (m == 15) ? w0 : 0.0f; }
            a[2 * j + h] = (_Float16)v;
        }
    }
    return a;
}

// ---------------------------------------------------------------------------
// Kernel 5: blur pass 2 (y axis), f32 in (d_out) -> f16 tmp. One wave per
// 16(y) x 16(z) tile at fixed x; single v_wmma_f32_16x16x32_f16 per tile.
// LDS tile stored TRANSPOSED as f32 [z=16][y=18] so each lane's 16 B halves
// come from one contiguous LDS run; staged via async global->LDS b32 copies.
// B layout (32x16, 16-bit): VGPR j: lanes 0-15 K=2j,2j+1 ; lanes 16-31 +16.
// ---------------------------------------------------------------------------
__global__ void __launch_bounds__(256)
k_blur_y(const float* __restrict__ in, _Float16* __restrict__ outh) {
    __shared__ float smemF[8 * 16 * 18];             // 9216B
    const int wave = threadIdx.x >> 5;
    const int lane = threadIdx.x & 31;
    const int tile = blockIdx.x * 8 + wave;          // 0..65535
    const int x  = tile >> 8;
    const int y0 = ((tile >> 4) & 15) << 4;
    const int z0 = (tile & 15) << 4;
    float* smF = smemF + wave * 288;

    // stage 18 y-rows (wrap halos at r=0,17) x 16 z, transposed into [c][r]
#if HAVE_ASYNC_LDS
    for (int t = lane; t < 288; t += 32) {
        const int r = t >> 4, c = t & 15;
        const int y = (y0 + r - 1) & 255;
        async_copy_b32(in + ((((x << 8) | y) << 8) + z0 + c),
                       smF + (c * 18 + r));
    }
    async_wait0();
#else
    for (int t = lane; t < 288; t += 32) {
        const int r = t >> 4, c = t & 15;
        const int y = (y0 + r - 1) & 255;
        smF[c * 18 + r] = in[((((x << 8) | y) << 8) + z0 + c)];
    }
#endif
    __syncthreads();

    float w0, w1; gauss_w(w0, w1);
    const v16h a = build_A(w0, w1);

    const int n  = lane & 15;
    const bool hi = (lane >= 16);
    const float* col = smF + n * 18;                 // y run for this z column
    v16h b;
    if (!hi) {                                       // K=0..15 -> y0..y0+15
        #pragma unroll
        for (int i = 0; i < 16; ++i) b[i] = (_Float16)col[1 + i];
    } else {                                         // K=16,17 -> halos
        #pragma unroll
        for (int i = 0; i < 16; ++i) b[i] = (_Float16)0.0f;
        b[0] = (_Float16)col[0];
        b[1] = (_Float16)col[17];
    }

    v8f c = {};
    const v8f d = __builtin_amdgcn_wmma_f32_16x16x32_f16(
        false, a, false, b, (short)0, c, false, false);

    const int mbase = (lane >> 4) << 3;              // C/D layout: M = r (+8 hi)
    #pragma unroll
    for (int r = 0; r < 8; ++r) {
        const int y = y0 + mbase + r;
        outh[(((x << 8) | y) << 8) | (z0 + n)] = (_Float16)d[r];
    }
}

// ---------------------------------------------------------------------------
// Kernel 6: blur pass 3 (z axis), f16 tmp -> f32 d_out. Conv axis = M (z),
// batch = N (16 y rows at fixed x). Same Toeplitz A. LDS rows padded to 20
// halves so the 16 center halves are dword-aligned for async b32 staging;
// wrap halo halves (cols 1 and 18) staged with plain DS stores.
// ---------------------------------------------------------------------------
__global__ void __launch_bounds__(256)
k_blur_z(const _Float16* __restrict__ in, float* __restrict__ out) {
    __shared__ _Float16 smem[8 * 16 * 20];           // 5120B
    const int wave = threadIdx.x >> 5;
    const int lane = threadIdx.x & 31;
    const int tile = blockIdx.x * 8 + wave;
    const int x  = tile >> 8;
    const int y0 = ((tile >> 4) & 15) << 4;
    const int z0 = (tile & 15) << 4;
    _Float16* sm = smem + wave * 320;

    // center 16x16 halves: 8 aligned dwords per row
#if HAVE_ASYNC_LDS
    for (int t = lane; t < 128; t += 32) {
        const int r = t >> 3, dq = t & 7;
        async_copy_b32(in + ((((x << 8) | (y0 + r)) << 8) + z0 + dq * 2),
                       sm + (r * 20 + 2 + dq * 2));
    }
#else
    for (int t = lane; t < 256; t += 32) {
        const int r = t >> 4, cz = t & 15;
        sm[r * 20 + 2 + cz] = in[(((x << 8) | (y0 + r)) << 8) + z0 + cz];
    }
#endif
    // wrap halos: col 1 = z0-1, col 18 = z0+16 (one half per lane)
    {
        const int r = lane & 15, side = lane >> 4;
        const int z = side ? ((z0 + 16) & 255) : ((z0 - 1) & 255);
        sm[r * 20 + (side ? 18 : 1)] =
            in[((((x << 8) | (y0 + r)) << 8) + z)];
    }
#if HAVE_ASYNC_LDS
    async_wait0();
#endif
    __syncthreads();

    float w0, w1; gauss_w(w0, w1);
    const v16h a = build_A(w0, w1);

    const int n  = lane & 15;                        // batch column -> y row
    const bool hi = (lane >= 16);
    const _Float16* row = sm + n * 20;
    v16h b;
    if (!hi) {                                       // K=0..15 -> z0..z0+15
        #pragma unroll
        for (int i = 0; i < 16; ++i) b[i] = row[2 + i];
    } else {                                         // K=16,17 -> halos
        #pragma unroll
        for (int i = 0; i < 16; ++i) b[i] = (_Float16)0.0f;
        b[0] = row[1];
        b[1] = row[18];
    }

    v8f c = {};
    const v8f d = __builtin_amdgcn_wmma_f32_16x16x32_f16(
        false, a, false, b, (short)0, c, false, false);

    const int mbase = (lane >> 4) << 3;              // conv row -> z
    #pragma unroll
    for (int r = 0; r < 8; ++r) {
        out[(((x << 8) | (y0 + n)) << 8) | (z0 + mbase + r)] = d[r];
    }
}

// ---------------------------------------------------------------------------
extern "C" void kernel_launch(void* const* d_in, const int* in_sizes, int n_in,
                              void* d_out, int out_size, void* d_ws, size_t ws_size,
                              hipStream_t stream) {
    const float* xyz = (const float*)d_in[0];
    const float* vdw = (const float*)d_in[1];
    const int n_atoms = in_sizes[1];
    float* out = (float*)d_out;

    char* ws = (char*)d_ws;
    uint32_t* P  = (uint32_t*)ws;        // 2MB protein bits
    uint32_t* B  = P  + WORDS;           // 2MB boundary bits (raw, then final)
    uint32_t* Z0 = B  + WORDS;           // 2MB def_solv
    uint32_t* Z1 = Z0 + WORDS;           // 2MB z-dilated r=1
    uint32_t* Z2 = Z1 + WORDS;           // 2MB z-dilated r=2
    uint32_t* H  = Z2 + WORDS;           // 2MB has_nb
    _Float16* T1 = (_Float16*)(ws + (size_t)6 * WORDS * 4);  // 32MB f16 tmp

    k_zero    <<<(2 * WORDS / 4 + 255) / 256, 256, 0, stream>>>((uint4*)P, 2 * WORDS / 4);
    k_scatter <<<n_atoms, 256, 0, stream>>>(xyz, vdw, P, B, n_atoms);
    k_finalize<<<WORDS / 256, 256, 0, stream>>>(P, B, Z0, Z1, Z2);
    k_dilate  <<<WORDS / 256, 256, 0, stream>>>(Z0, Z1, Z2, H);
    k_blur_x  <<<WORDS / 256, 256, 0, stream>>>(P, B, H, out);
    k_blur_y  <<<65536 / 8, 256, 0, stream>>>(out, T1);
    k_blur_z  <<<65536 / 8, 256, 0, stream>>>(T1, out);
}
